// CausalSelfAttention_67156108640705
// MI455X (gfx1250) — matrix operand order repaired and checked by reading, not compiled, not run on hardware
//
#include <hip/hip_runtime.h>

// ---------------- problem constants ----------------
constexpr int NB = 2;      // batch
constexpr int NT = 2048;   // sequence
constexpr int NC = 1024;   // channels
constexpr int NH = 16;     // heads
constexpr int ND = 64;     // head dim

// ---------------- vector types ----------------
typedef __attribute__((ext_vector_type(16))) __bf16        v16bf;
typedef __attribute__((ext_vector_type(8)))  float         v8f;
typedef __attribute__((ext_vector_type(4)))  unsigned int  v4u;
typedef int g4i __attribute__((vector_size(16)));  // matches async-LDS builtin param

union V16 { v16bf v; v4u q[2]; };
union U8V { v4u q; __bf16 b[8]; };

// ---- CDNA5 async global->LDS path (guarded; falls back to sync copy) ----
#if defined(__has_builtin)
#if __has_builtin(__builtin_amdgcn_global_load_async_to_lds_b128) && \
    __has_builtin(__builtin_amdgcn_s_wait_asynccnt)
#define USE_ASYNC_LDS 1
#endif
#endif
#ifndef USE_ASYNC_LDS
#define USE_ASYNC_LDS 0
#endif

// copy 16B (8 bf16) global -> LDS
static __device__ __forceinline__ void cp16(__bf16* lds_dst, const __bf16* g_src) {
#if USE_ASYNC_LDS
    __builtin_amdgcn_global_load_async_to_lds_b128(
        (__attribute__((address_space(1))) g4i*)(g_src),
        (__attribute__((address_space(3))) g4i*)(lds_dst), 0, 0);
#else
    *(v4u*)lds_dst = *(const v4u*)g_src;
#endif
}
#if USE_ASYNC_LDS
#define ASYNC_WAIT(n) __builtin_amdgcn_s_wait_asynccnt(n)
#else
#define ASYNC_WAIT(n)
#endif

// fp32 -> bf16, round-to-nearest-even (pure bit ops; no __bf16 arithmetic)
static __device__ __forceinline__ __bf16 f2bf(float f) {
    unsigned u = __float_as_uint(f);
    u += 0x7FFFu + ((u >> 16) & 1u);
    unsigned short h = (unsigned short)(u >> 16);
    return __builtin_bit_cast(__bf16, h);
}

// A-fragment (16x32 bf16, MxK): lane holds row m=l&15;
// elements 0..7 -> k = 8*hg + e, elements 8..15 -> k = 16 + 8*hg + (e-8)
static __device__ __forceinline__ v16bf load_frag_a(const __bf16* rowbase, int hg) {
    V16 u;
    u.q[0] = *(const v4u*)(rowbase + 8 * hg);
    u.q[1] = *(const v4u*)(rowbase + 16 + 8 * hg);
    return u.v;
}
// B-fragment (32x16 bf16, KxN): lane holds column n=l&15;
// elements 0..7 -> k = 8*hg + e, elements 8..15 -> k = 16 + 8*hg + (e-8)
static __device__ __forceinline__ v16bf load_frag_b(const __bf16* colbase, int hg) {
    V16 u;
    u.q[0] = *(const v4u*)(colbase + 8 * hg);
    u.q[1] = *(const v4u*)(colbase + 16 + 8 * hg);
    return u.v;
}

// ---------------- fp32 -> bf16 convert ----------------
__global__ void cvt_f32_bf16(const float* __restrict__ in, __bf16* __restrict__ out, int n) {
    int i = blockIdx.x * blockDim.x + threadIdx.x;
    int stride = gridDim.x * blockDim.x;
    for (; i < n; i += stride) out[i] = f2bf(in[i]);
}

// ---------------- GEMM: out = A @ W^T + bias ----------------
// A: [M,K] bf16 row-major, W: [N,K] bf16 row-major.
// MODE 0: write bf16 to head layout [B,H,T,64];  MODE 1: write fp32 [M,N].
// Block tile 128x128, 256 threads (8 wave32), wave tile 32x64,
// K-step 64, double-buffered LDS (2 x (16KB A + 16KB B) = 64KB).
template <int MODE>
__global__ __launch_bounds__(256)
void gemm_bf16(const __bf16* __restrict__ A, const __bf16* __restrict__ W,
               const float* __restrict__ bias, void* __restrict__ outp,
               int M, int N, int K) {
    __shared__ __bf16 As[2][128 * 64];
    __shared__ __bf16 Bs[2][128 * 64];

    const int tid  = threadIdx.x;
    const int lane = tid & 31;
    const int wv   = tid >> 5;
    const int hg   = lane >> 4;
    const int ln   = lane & 15;
    const int wm   = wv & 3;   // 4 M-subtiles of 32
    const int wn   = wv >> 2;  // 2 N-subtiles of 64
    const int m0   = blockIdx.y * 128;
    const int n0   = blockIdx.x * 128;

    v8f zero = {};
    v8f acc[2][4];
#pragma unroll
    for (int i = 0; i < 2; ++i)
#pragma unroll
        for (int j = 0; j < 4; ++j) acc[i][j] = zero;

    // stage one 128x64 A tile + 128x64 B tile (4 x 16B per thread per matrix,
    // 8 async ops per thread per stage)
    auto stage = [&](int buf, int kk) {
#pragma unroll
        for (int i = 0; i < 4; ++i) {
            int c   = tid + i * 256;   // 0..1023
            int r   = c >> 3;          // 0..127
            int col = (c & 7) * 8;     // 0..56
            cp16(&As[buf][r * 64 + col], A + (size_t)(m0 + r) * K + kk + col);
            cp16(&Bs[buf][r * 64 + col], W + (size_t)(n0 + r) * K + kk + col);
        }
    };

    const int nk = K >> 6;  // K/64 steps
    stage(0, 0);
    for (int k = 0; k < nk; ++k) {
        const int cur = k & 1;
        if (k + 1 < nk) {
            stage(1 - cur, (k + 1) << 6);
            ASYNC_WAIT(8);  // oldest stage (cur) retired (in-order ASYNCcnt)
        } else {
            ASYNC_WAIT(0);
        }
        __syncthreads();

#pragma unroll
        for (int kc = 0; kc < 2; ++kc) {
            v16bf aF[2], bF[4];
#pragma unroll
            for (int mt = 0; mt < 2; ++mt)
                aF[mt] = load_frag_a(&As[cur][(wm * 32 + mt * 16 + ln) * 64 + kc * 32], hg);
#pragma unroll
            for (int nt = 0; nt < 4; ++nt)
                bF[nt] = load_frag_b(&Bs[cur][(wn * 64 + nt * 16 + ln) * 64 + kc * 32], hg);
#pragma unroll
            for (int mt = 0; mt < 2; ++mt)
#pragma unroll
                for (int nt = 0; nt < 4; ++nt)
                    acc[mt][nt] = __builtin_amdgcn_wmma_f32_16x16x32_bf16(
                        false, aF[mt], false, bF[nt], (short)0, acc[mt][nt], false, false);
        }
        __syncthreads();
    }

    // epilogue: +bias, scatter per C-layout (element r -> M = 8*hg + r, N = ln)
#pragma unroll
    for (int mt = 0; mt < 2; ++mt) {
#pragma unroll
        for (int nt = 0; nt < 4; ++nt) {
            const int gn = n0 + wn * 64 + nt * 16 + ln;
            const float bv = bias[gn];
#pragma unroll
            for (int r = 0; r < 8; ++r) {
                const int gm = m0 + wm * 32 + mt * 16 + 8 * hg + r;
                const float v = acc[mt][nt][r] + bv;
                if (MODE == 0) {
                    const int hh = gn >> 6, dd = gn & 63;
                    const int bb = gm >> 11, tt = gm & 2047;  // M = B*T, T = 2048
                    ((__bf16*)outp)[(((size_t)(bb * NH + hh) * NT + tt) << 6) + dd] = f2bf(v);
                } else {
                    ((float*)outp)[(size_t)gm * N + gn] = v;
                }
            }
        }
    }
}

// ---------------- causal flash attention ----------------
// grid (T/64, H, B), 128 threads = 4 wave32. Each wave owns 16 query rows.
// Q,K,V: [B,H,T,64] bf16. Output Y: [B,T,C] bf16.
__global__ __launch_bounds__(128)
void flash_attn(const __bf16* __restrict__ Qp, const __bf16* __restrict__ Kp,
                const __bf16* __restrict__ Vp, __bf16* __restrict__ Y) {
    __shared__ __bf16 Kt[64 * 64];      // [key][d]  (B-frag reads contiguous in d)
    __shared__ __bf16 Vt[64 * 72];      // [d][key], padded stride 72 (B-frag contiguous in key)
    __shared__ __bf16 Pt[4 * 16 * 64];  // per-wave P tile [q][key]

    const int tid  = threadIdx.x;
    const int lane = tid & 31;
    const int wv   = tid >> 5;
    const int hg   = lane >> 4;
    const int ln   = lane & 15;
    const int q0   = blockIdx.x * 64;
    const int hh   = blockIdx.y;
    const int bb   = blockIdx.z;

    const size_t hb = ((size_t)(bb * NH + hh)) * NT * ND;
    const __bf16* Qh = Qp + hb;
    const __bf16* Kh = Kp + hb;
    const __bf16* Vh = Vp + hb;

    // Q fragments for this wave's 16 rows (kept in registers across the key loop)
    const int qrow = q0 + wv * 16 + ln;
    v16bf aQ[2];
#pragma unroll
    for (int kc = 0; kc < 2; ++kc)
        aQ[kc] = load_frag_a(Qh + (size_t)qrow * ND + kc * 32, hg);

    v8f zero = {};
    v8f o[4];
#pragma unroll
    for (int nt = 0; nt < 4; ++nt) o[nt] = zero;
    float mrow[8], lrow[8];
#pragma unroll
    for (int r = 0; r < 8; ++r) { mrow[r] = -1e30f; lrow[r] = 0.f; }

    const int nkt = q0 / 64 + 1;  // causal: key tiles 0 .. q0/64
    for (int kt = 0; kt < nkt; ++kt) {
        const int k0 = kt * 64;
        __syncthreads();  // previous tile fully consumed
        // stage K (async, row-major) and V (manual transpose) tiles: 8KB each
#pragma unroll
        for (int i = 0; i < 4; ++i) {
            int c  = tid + i * 128;   // 0..511
            int r  = c >> 3;          // key row 0..63
            int d0 = (c & 7) * 8;     // d 0,8,...,56
            cp16(Kt + r * 64 + d0, Kh + (size_t)(k0 + r) * ND + d0);
            U8V vv; vv.q = *(const v4u*)(Vh + (size_t)(k0 + r) * ND + d0);
#pragma unroll
            for (int j = 0; j < 8; ++j) Vt[(d0 + j) * 72 + r] = vv.b[j];
        }
        ASYNC_WAIT(0);
        __syncthreads();

        // S = (Q K^T) * scale  — 4 N-tiles of 16 keys, K-dim = 64 (2 chunks)
        v8f s[4];
#pragma unroll
        for (int nt = 0; nt < 4; ++nt) {
            s[nt] = zero;
#pragma unroll
            for (int kc = 0; kc < 2; ++kc) {
                v16bf bK = load_frag_b(Kt + (nt * 16 + ln) * 64 + kc * 32, hg);
                s[nt] = __builtin_amdgcn_wmma_f32_16x16x32_bf16(
                    false, aQ[kc], false, bK, (short)0, s[nt], false, false);
            }
        }
        // scale + causal mask (element r: q = q0+16w+8*hg+r, key = k0+16nt+ln)
#pragma unroll
        for (int nt = 0; nt < 4; ++nt) {
            int key = k0 + nt * 16 + ln;
#pragma unroll
            for (int r = 0; r < 8; ++r) {
                int qi   = q0 + wv * 16 + 8 * hg + r;
                float sv = s[nt][r] * 0.125f;  // 1/sqrt(64)
                s[nt][r] = (key <= qi) ? sv : -1e30f;
            }
        }
        // online softmax: rows live in 16-lane N-groups
#pragma unroll
        for (int r = 0; r < 8; ++r) {
            float cur = fmaxf(fmaxf(s[0][r], s[1][r]), fmaxf(s[2][r], s[3][r]));
#pragma unroll
            for (int off = 1; off < 16; off <<= 1)
                cur = fmaxf(cur, __shfl_xor(cur, off, 16));
            float mn    = fmaxf(mrow[r], cur);
            float alpha = __expf(mrow[r] - mn);
            mrow[r]     = mn;
            float rs = 0.f;
#pragma unroll
            for (int nt = 0; nt < 4; ++nt) {
                float p  = __expf(s[nt][r] - mn);
                s[nt][r] = p;
                rs += p;
            }
#pragma unroll
            for (int off = 1; off < 16; off <<= 1)
                rs += __shfl_xor(rs, off, 16);
            lrow[r] = lrow[r] * alpha + rs;
#pragma unroll
            for (int nt = 0; nt < 4; ++nt) o[nt][r] = o[nt][r] * alpha;
        }
        // bounce P through LDS: C-layout -> row-major [q][key] -> A-fragment layout
#pragma unroll
        for (int nt = 0; nt < 4; ++nt)
#pragma unroll
            for (int r = 0; r < 8; ++r)
                Pt[wv * 1024 + (8 * hg + r) * 64 + nt * 16 + ln] = f2bf(s[nt][r]);
        __syncthreads();

        v16bf aP[2];
#pragma unroll
        for (int kc = 0; kc < 2; ++kc)
            aP[kc] = load_frag_a(Pt + wv * 1024 + ln * 64 + kc * 32, hg);
#pragma unroll
        for (int nt = 0; nt < 4; ++nt)
#pragma unroll
            for (int kc = 0; kc < 2; ++kc) {
                v16bf bV = load_frag_b(Vt + (nt * 16 + ln) * 72 + kc * 32, hg);
                o[nt] = __builtin_amdgcn_wmma_f32_16x16x32_bf16(
                    false, aP[kc], false, bV, (short)0, o[nt], false, false);
            }
    }

    // normalize and write Y[b, q, h*64 + d] as bf16
#pragma unroll
    for (int nt = 0; nt < 4; ++nt) {
        int col = hh * 64 + nt * 16 + ln;
#pragma unroll
        for (int r = 0; r < 8; ++r) {
            int qi = q0 + wv * 16 + 8 * hg + r;
            Y[(size_t)(bb * NT + qi) * NC + col] = f2bf(o[nt][r] * (1.f / lrow[r]));
        }
    }
}

// ---------------- launch ----------------
extern "C" void kernel_launch(void* const* d_in, const int* in_sizes, int n_in,
                              void* d_out, int out_size, void* d_ws, size_t ws_size,
                              hipStream_t stream) {
    const float* x  = (const float*)d_in[0];
    const float* Wq = (const float*)d_in[1];
    const float* bq = (const float*)d_in[2];
    const float* Wk = (const float*)d_in[3];
    const float* bk = (const float*)d_in[4];
    const float* Wv = (const float*)d_in[5];
    const float* bv = (const float*)d_in[6];
    const float* Wp = (const float*)d_in[7];
    const float* bp = (const float*)d_in[8];

    char* ws = (char*)d_ws;
    const size_t MB = (size_t)1 << 20;
    __bf16* xb  = (__bf16*)(ws);            //  8 MB: x bf16 [B*T, C]
    __bf16* Wqb = (__bf16*)(ws + 8 * MB);   //  2 MB
    __bf16* Wkb = (__bf16*)(ws + 10 * MB);  //  2 MB
    __bf16* Wvb = (__bf16*)(ws + 12 * MB);  //  2 MB
    __bf16* Wpb = (__bf16*)(ws + 14 * MB);  //  2 MB
    __bf16* Qh  = (__bf16*)(ws + 16 * MB);  //  8 MB: [B,H,T,64]
    __bf16* Kh  = (__bf16*)(ws + 24 * MB);  //  8 MB
    __bf16* Vh  = (__bf16*)(ws + 32 * MB);  //  8 MB
    __bf16* Yb  = (__bf16*)(ws + 40 * MB);  //  8 MB: attention out [B*T, C]

    const int nx = NB * NT * NC;  // 4,194,304
    const int nw = NC * NC;       // 1,048,576
    cvt_f32_bf16<<<2048, 256, 0, stream>>>(x,  xb,  nx);
    cvt_f32_bf16<<<1024, 256, 0, stream>>>(Wq, Wqb, nw);
    cvt_f32_bf16<<<1024, 256, 0, stream>>>(Wk, Wkb, nw);
    cvt_f32_bf16<<<1024, 256, 0, stream>>>(Wv, Wvb, nw);
    cvt_f32_bf16<<<1024, 256, 0, stream>>>(Wp, Wpb, nw);

    dim3 gg(NC / 128, (NB * NT) / 128);  // (8, 32)
    gemm_bf16<0><<<gg, 256, 0, stream>>>(xb, Wqb, bq, Qh, NB * NT, NC, NC);
    gemm_bf16<0><<<gg, 256, 0, stream>>>(xb, Wkb, bk, Kh, NB * NT, NC, NC);
    gemm_bf16<0><<<gg, 256, 0, stream>>>(xb, Wvb, bv, Vh, NB * NT, NC, NC);

    dim3 fg(NT / 64, NH, NB);  // (32, 16, 2)
    flash_attn<<<fg, 128, 0, stream>>>(Qh, Kh, Vh, Yb);

    gemm_bf16<1><<<gg, 256, 0, stream>>>(Yb, Wpb, bp, d_out, NB * NT, NC, NC);
}
